// AF_2_54022098649752
// MI455X (gfx1250) — matrix operand order repaired and checked by reading, not compiled, not run on hardware
//
#include <hip/hip_runtime.h>
#include <stdint.h>

#define BB    4096
#define NBR   64
#define DIM   512
#define INTER 128
#define EPSV  1e-5f
#define MT    32      // rows of B per workgroup
#define BLOCK 256     // 8 waves (wave32)

typedef __attribute__((ext_vector_type(16))) _Float16 v16h;
typedef __attribute__((ext_vector_type(8)))  _Float16 v8h;
typedef __attribute__((ext_vector_type(8)))  float    v8f;
typedef __attribute__((ext_vector_type(4)))  unsigned int v4u;
typedef __attribute__((ext_vector_type(4)))  int      v4i;
typedef __attribute__((ext_vector_type(8)))  int      v8i;

// ---------------- BN fold: c1[n,i], c2[n,d] ----------------
__global__ void bn_fold_kernel(const float* __restrict__ b1, const float* __restrict__ g1,
                               const float* __restrict__ beta1, const float* __restrict__ m1,
                               const float* __restrict__ v1,
                               const float* __restrict__ b2, const float* __restrict__ g2,
                               const float* __restrict__ beta2, const float* __restrict__ m2,
                               const float* __restrict__ v2,
                               float* __restrict__ c1, float* __restrict__ c2) {
  int idx = blockIdx.x * blockDim.x + threadIdx.x;   // over NBR*DIM
  if (idx >= NBR * DIM) return;
  int n = idx >> 9, d = idx & 511;
  float a2 = g2[idx] * rsqrtf(v2[idx] + EPSV);
  c2[idx] = (b2[idx] - m2[idx]) * a2 + beta2[idx];
  if (d < INTER) {
    int j = n * INTER + d;
    float a1 = g1[j] * rsqrtf(v1[j] + EPSV);
    c1[j] = (b1[j] - m1[j]) * a1 + beta1[j];
  }
}

// ---------------- fold BN scale into weights, f32 -> f16 ----------------
__global__ void fold_w1_kernel(const float* __restrict__ W1, const float* __restrict__ g1,
                               const float* __restrict__ v1, _Float16* __restrict__ w1h) {
  int idx = blockIdx.x * blockDim.x + threadIdx.x;   // NBR*INTER*DIM
  if (idx >= NBR * INTER * DIM) return;
  int ni = idx >> 9;                                 // n*INTER + i
  float a1 = g1[ni] * rsqrtf(v1[ni] + EPSV);
  w1h[idx] = (_Float16)(W1[idx] * a1);
}

__global__ void fold_w2_kernel(const float* __restrict__ W2, const float* __restrict__ g2,
                               const float* __restrict__ v2, _Float16* __restrict__ w2h) {
  int idx = blockIdx.x * blockDim.x + threadIdx.x;   // NBR*DIM*INTER
  if (idx >= NBR * DIM * INTER) return;
  int nd = idx >> 7;                                 // n*DIM + d
  float a2 = g2[nd] * rsqrtf(v2[nd] + EPSV);
  w2h[idx] = (_Float16)(W2[idx] * a2);
}

// ---------------- s = sum_n t[:,n,:]  -> f16 ----------------
__global__ void sum_kernel(const float* __restrict__ t, _Float16* __restrict__ s16) {
  int idx = blockIdx.x * blockDim.x + threadIdx.x;   // BB*DIM
  if (idx >= BB * DIM) return;
  int b = idx >> 9, d = idx & 511;
  const float* p = t + (size_t)b * NBR * DIM + d;
  float acc = 0.f;
#pragma unroll 8
  for (int n = 0; n < NBR; ++n) acc += p[(size_t)n * DIM];
  s16[idx] = (_Float16)acc;
}

// ---------------- fused per-branch GEMM chain + weighted reduce ----------------
__launch_bounds__(BLOCK)
__global__ void fused_kernel(const float* __restrict__ t,
                             const _Float16* __restrict__ w1h,
                             const _Float16* __restrict__ w2h,
                             const float* __restrict__ c1,
                             const float* __restrict__ c2,
                             const _Float16* __restrict__ s16,
                             float* __restrict__ out) {
  __shared__ _Float16 sbuf[MT][DIM + 8];     // padded: stride 1040B (conflict-free b128 reads)
  __shared__ _Float16 hbuf[MT][INTER + 8];   // padded: stride 272B

  const int tid  = threadIdx.x;
  const int wave = tid >> 5;                 // 0..7
  const int lane = tid & 31;
  const int lr   = lane & 15;                // row/col-in-tile
  const int hs   = lane >> 4;                // half-wave select
  const int b0   = blockIdx.x * MT;

  // ---- stage this WG's 32x512 f16 s-tile into LDS (read once, reused x64 branches)
#if defined(__gfx1250__) && __has_builtin(__builtin_amdgcn_tensor_load_to_lds)
  // Tensor Data Mover: 2D tile 32 rows x 512 f16, with LDS padding of 4 DWORDs
  // (16B) after every 256 DWORDs (one 1024B row) -> exactly sbuf's 520-half stride.
  if (wave == 0) {
    uint64_t gaddr  = (uint64_t)(uintptr_t)(s16 + (size_t)b0 * DIM);
    uint32_t ldsoff = (uint32_t)(uintptr_t)(&sbuf[0][0]);
    v4u g0;
    g0.x = 1u;                                          // count=1 (valid user D#)
    g0.y = ldsoff;                                      // lds_addr
    g0.z = (uint32_t)gaddr;                             // global_addr[31:0]
    g0.w = ((uint32_t)(gaddr >> 32) & 0x01FFFFFFu) | (2u << 30);  // addr[56:32], type=2
    v8i g1;
    g1[0] = (int)((1u << 16) |                          // data_size = 2 bytes
                  (1u << 20) |                          // pad_enable
                  (7u << 22) |                          // pad_interval: 256 DWORDs
                  (3u << 25));                          // pad_amount: 4 DWORDs
    g1[1] = (int)(512u << 16);                          // tensor_dim0 = 512
    g1[2] = (int)(32u << 16);                           // tensor_dim1 = 32
    g1[3] = (int)(512u << 16);                          // tile_dim0 = 512
    g1[4] = 32;                                         // tile_dim1 = 32
    g1[5] = 512;                                        // tensor_dim0_stride = 512
    g1[6] = 0;
    g1[7] = 0;
    v4i z4 = {0, 0, 0, 0};
#if __clang_major__ >= 23
    v8i z8 = {0, 0, 0, 0, 0, 0, 0, 0};
    __builtin_amdgcn_tensor_load_to_lds(g0, g1, z4, z4, z8, 0);
#else
    __builtin_amdgcn_tensor_load_to_lds(g0, g1, z4, z4, 0);
#endif
    __builtin_amdgcn_s_wait_tensorcnt(0);
  }
  __syncthreads();
#else
  {
    int row  = tid >> 3;                     // 0..31
    int colb = (tid & 7) * 64;
    const _Float16* src = s16 + (size_t)(b0 + row) * DIM + colb;
#pragma unroll
    for (int c = 0; c < 64; c += 8)
      *(v8h*)&sbuf[row][colb + c] = *(const v8h*)(src + c);
  }
  __syncthreads();
#endif

  float out_acc[8][8];
#pragma unroll
  for (int i = 0; i < 8; ++i)
#pragma unroll
    for (int r = 0; r < 8; ++r) out_acc[i][r] = 0.f;

  for (int n = 0; n < NBR; ++n) {
    // ================= GEMM1: h(32x128) = relu(s @ W1'^T + c1) =================
    const int nt = wave;                     // one 16-wide N tile per wave
    v8f acc0 = {};
    v8f acc1 = {};
    const _Float16* bbase = w1h + (size_t)(n * INTER + nt * 16 + lr) * DIM + hs * 16;
#pragma unroll 4
    for (int kk = 0; kk < 16; ++kk) {
      v16h bmat = *(const v16h*)(bbase + kk * 32);   // 32B contiguous (B layout)
      const int col0 = kk * 32 + hs * 16;            // A layout K placement
      v16h a0, a1;
      {
        v8h lo = *(const v8h*)&sbuf[lr][col0];
        v8h hi = *(const v8h*)&sbuf[lr][col0 + 8];
#pragma unroll
        for (int e = 0; e < 8; ++e) { a0[e] = lo[e]; a0[e + 8] = hi[e]; }
      }
      {
        v8h lo = *(const v8h*)&sbuf[16 + lr][col0];
        v8h hi = *(const v8h*)&sbuf[16 + lr][col0 + 8];
#pragma unroll
        for (int e = 0; e < 8; ++e) { a1[e] = lo[e]; a1[e + 8] = hi[e]; }
      }
      acc0 = __builtin_amdgcn_wmma_f32_16x16x32_f16(false, a0, false, bmat, (short)0, acc0, false, false);
      acc1 = __builtin_amdgcn_wmma_f32_16x16x32_f16(false, a1, false, bmat, (short)0, acc1, false, false);
    }
    // epilogue: +c1, relu, f16 -> hbuf (C layout: vgpr r = row r+8*hs, col = lr)
    {
      float c1v = c1[n * INTER + nt * 16 + lr];
#pragma unroll
      for (int r = 0; r < 8; ++r) {
        float h0 = acc0[r] + c1v; h0 = h0 > 0.f ? h0 : 0.f;
        float h1 = acc1[r] + c1v; h1 = h1 > 0.f ? h1 : 0.f;
        hbuf[r + hs * 8][nt * 16 + lr]      = (_Float16)h0;
        hbuf[16 + r + hs * 8][nt * 16 + lr] = (_Float16)h1;
      }
    }
    // prefetch next branch's W1' toward cache while we sync/compute GEMM2
    if (n + 1 < NBR)
      __builtin_prefetch(w1h + (size_t)(n + 1) * INTER * DIM + (size_t)tid * 256, 0, 1);
    __syncthreads();

    // ================= GEMM2 + sigmoid*t accumulate =================
#pragma unroll
    for (int q = 0; q < 4; ++q) {
      const int d0  = (wave + 8 * q) * 16;
      const float c2v = c2[n * DIM + d0 + lr];
#pragma unroll
      for (int mt = 0; mt < 2; ++mt) {
        v8f yacc = {};
        const _Float16* b2base = w2h + (size_t)(n * DIM + d0 + lr) * INTER + hs * 16;
#pragma unroll
        for (int k = 0; k < 4; ++k) {
          v16h bmat = *(const v16h*)(b2base + k * 32);
          const int col0 = k * 32 + hs * 16;
          v16h amat;
          v8h lo = *(const v8h*)&hbuf[mt * 16 + lr][col0];
          v8h hi = *(const v8h*)&hbuf[mt * 16 + lr][col0 + 8];
#pragma unroll
          for (int e = 0; e < 8; ++e) { amat[e] = lo[e]; amat[e + 8] = hi[e]; }
          yacc = __builtin_amdgcn_wmma_f32_16x16x32_f16(false, amat, false, bmat, (short)0, yacc, false, false);
        }
        // epilogue directly in C layout: each half-wave reads 64B-contiguous t
        const float* tbase = t + ((size_t)(b0 + mt * 16 + hs * 8) * NBR + n) * DIM + d0 + lr;
#pragma unroll
        for (int r = 0; r < 8; ++r) {
          float y  = yacc[r] + c2v;
          float wv = 1.f / (1.f + __expf(-y));
          float tv = tbase[(size_t)r * NBR * DIM];
          out_acc[q * 2 + mt][r] += wv * tv;
        }
      }
    }
    __syncthreads();   // hbuf consumed; safe to overwrite next branch
  }

  // ---- write out (each (b,d) of the 32x512 tile owned by exactly one lane-reg)
#pragma unroll
  for (int q = 0; q < 4; ++q) {
    const int d0 = (wave + 8 * q) * 16;
#pragma unroll
    for (int mt = 0; mt < 2; ++mt) {
      float* obase = out + (size_t)(b0 + mt * 16 + hs * 8) * DIM + d0 + lr;
#pragma unroll
      for (int r = 0; r < 8; ++r)
        obase[(size_t)r * DIM] = 3.f * out_acc[q * 2 + mt][r];
    }
  }
}

extern "C" void kernel_launch(void* const* d_in, const int* in_sizes, int n_in,
                              void* d_out, int out_size, void* d_ws, size_t ws_size,
                              hipStream_t stream) {
  const float* t     = (const float*)d_in[0];
  const float* W1    = (const float*)d_in[1];
  const float* b1    = (const float*)d_in[2];
  const float* g1    = (const float*)d_in[3];
  const float* beta1 = (const float*)d_in[4];
  const float* m1    = (const float*)d_in[5];
  const float* v1    = (const float*)d_in[6];
  const float* W2    = (const float*)d_in[7];
  const float* b2    = (const float*)d_in[8];
  const float* g2    = (const float*)d_in[9];
  const float* beta2 = (const float*)d_in[10];
  const float* m2    = (const float*)d_in[11];
  const float* v2    = (const float*)d_in[12];
  float* out = (float*)d_out;

  char* ws = (char*)d_ws;
  _Float16* w1h = (_Float16*)ws; ws += (size_t)NBR * INTER * DIM * 2;  // 8.39 MB
  _Float16* w2h = (_Float16*)ws; ws += (size_t)NBR * DIM * INTER * 2;  // 8.39 MB
  _Float16* s16 = (_Float16*)ws; ws += (size_t)BB * DIM * 2;           // 4.19 MB
  float*    c1  = (float*)ws;    ws += (size_t)NBR * INTER * 4;
  float*    c2  = (float*)ws;    ws += (size_t)NBR * DIM * 4;

  bn_fold_kernel<<<(NBR * DIM + 255) / 256, 256, 0, stream>>>(
      b1, g1, beta1, m1, v1, b2, g2, beta2, m2, v2, c1, c2);
  fold_w1_kernel<<<(NBR * INTER * DIM + 255) / 256, 256, 0, stream>>>(W1, g1, v1, w1h);
  fold_w2_kernel<<<(NBR * DIM * INTER + 255) / 256, 256, 0, stream>>>(W2, g2, v2, w2h);
  sum_kernel<<<(BB * DIM + 255) / 256, 256, 0, stream>>>(t, s16);
  fused_kernel<<<BB / MT, BLOCK, 0, stream>>>(t, w1h, w2h, c1, c2, s16, out);
}
